// SelfAttnLikeModule_65283502899242
// MI455X (gfx1250) — matrix operand order repaired and checked by reading, not compiled, not run on hardware
//
#include <hip/hip_runtime.h>
#include <hip/hip_bf16.h>
#include <math.h>

typedef __attribute__((ext_vector_type(16))) _Float16 v16h;
typedef __attribute__((ext_vector_type(8)))  float    v8f;
typedef _Float16 h16;

#define B_  2
#define S_  2048
#define D_  1024
#define H_  16
#define DH_ 64

#define USE_ASYNC 1

static __device__ __forceinline__ v8f wmma_f16(v16h a, v16h b, v8f c) {
  // D = A(16x32 f16) * B(32x16 f16) + C(16x16 f32)
  return __builtin_amdgcn_wmma_f32_16x16x32_f16(
      /*neg_a=*/false, a, /*neg_b=*/false, b,
      /*c_mod=*/(short)0, c, /*reuse_a=*/false, /*reuse_b=*/false);
}

// 16B global -> LDS copy. Async (CDNA5 GLOBAL_LOAD_ASYNC_TO_LDS_B128,
// tracked by ASYNCcnt) or plain fallback.
static __device__ __forceinline__ void cp16_async(h16* dst_lds, const h16* src_g) {
#if USE_ASYNC
  unsigned lds = (unsigned)(size_t)dst_lds;   // low 32 bits = wave LDS offset
  asm volatile("global_load_async_to_lds_b128 %0, %1, off"
               :: "v"(lds), "v"(src_g) : "memory");
#else
  *(uint4*)dst_lds = *(const uint4*)src_g;
#endif
}

static __device__ __forceinline__ void wait_async_all() {
#if USE_ASYNC
  asm volatile("s_wait_asynccnt 0x0" ::: "memory");
#endif
}

// ---------------------------------------------------------------- fp32 -> f16
__global__ void k_cvt(const float* __restrict__ in, h16* __restrict__ out, int n) {
  int i = blockIdx.x * blockDim.x + threadIdx.x;
  int stride = gridDim.x * blockDim.x;
  for (; i < n; i += stride) out[i] = (h16)in[i];
}

// ---------------------------------------------------------------- tiled GEMM
// A: [M=4096, K=1024] f16 row-major.  W: [K=1024, N=1024] f16 row-major.
// MODE 0: out16 = A*W written as [B,H,S,DH] f16 (head split)
// MODE 1: out32 = A*W + bias written as [B,S,D] f32
template<int MODE>
__global__ __launch_bounds__(256)
void k_gemm(const h16* __restrict__ A, const h16* __restrict__ W,
            h16* __restrict__ out16, float* __restrict__ out32,
            const float* __restrict__ bias) {
  constexpr int LDA = 48;  // halfs; row stride 96B (16B aligned)
  constexpr int LDB = 48;
  __shared__ h16 sA[2][128 * LDA];   // [m][k], double buffered
  __shared__ h16 sB[2][64  * LDB];   // [n][k]  (transposed tile of W)

  const int tid  = threadIdx.x;
  const int lane = tid & 31;
  const int w    = tid >> 5;
  const int hl   = lane >> 4;     // lane half (0/1)
  const int ln   = lane & 15;

  const int nTilesN = D_ / 64;                    // 16
  const int m0 = (int)(blockIdx.x / nTilesN) * 128;
  const int n0 = (int)(blockIdx.x % nTilesN) * 64;
  const int wm = w >> 1, wn = w & 1;              // 4x2 wave grid

  // staging coordinates
  const int ar = tid >> 2, akg = (tid & 3) << 3;          // A tile rows/cols
  const int bn = tid & 63, bkg = (tid >> 6) << 3;         // B tile n/k-group

  v8f acc[2][2];
  #pragma unroll
  for (int i = 0; i < 2; ++i)
    #pragma unroll
    for (int j = 0; j < 2; ++j)
      #pragma unroll
      for (int r = 0; r < 8; ++r) acc[i][j][r] = 0.f;

  auto stage = [&](int buf, int k0) {
    // A tile 128x32: two async b128 per thread
    cp16_async(&sA[buf][ar * LDA + akg],
               A + (size_t)(m0 + ar) * D_ + k0 + akg);
    cp16_async(&sA[buf][(ar + 64) * LDA + akg],
               A + (size_t)(m0 + ar + 64) * D_ + k0 + akg);
    // B tile 64x32 transposed into [n][k] (sync gather + one LDS b128 store)
    alignas(16) h16 tmp[8];
    #pragma unroll
    for (int j = 0; j < 8; ++j)
      tmp[j] = W[(size_t)(k0 + bkg + j) * D_ + n0 + bn];
    *(uint4*)(&sB[buf][bn * LDB + bkg]) = *(const uint4*)tmp;
  };

  stage(0, 0);
  wait_async_all();
  __syncthreads();

  constexpr int KSTEPS = D_ / 32;
  for (int ks = 0; ks < KSTEPS; ++ks) {
    const int buf = ks & 1;
    if (ks + 1 < KSTEPS) stage(buf ^ 1, (ks + 1) * 32);
    if (ks + 2 < KSTEPS)
      __builtin_prefetch(W + (size_t)((ks + 2) * 32 + bkg) * D_ + n0 + bn, 0, 1);

    // ---- fragments + 4 WMMAs
    v16h af[2], bf[2];
    #pragma unroll
    for (int mi = 0; mi < 2; ++mi) {
      const h16* arow = &sA[buf][(wm * 32 + mi * 16 + ln) * LDA];
      #pragma unroll
      for (int v = 0; v < 8; ++v) {
        int k = ((v >> 2) << 4) + (hl << 3) + ((v & 3) << 1);
        af[mi][2 * v]     = arow[k];
        af[mi][2 * v + 1] = arow[k + 1];
      }
    }
    #pragma unroll
    for (int ni = 0; ni < 2; ++ni) {
      const h16* bcol = &sB[buf][(wn * 32 + ni * 16 + ln) * LDB];
      #pragma unroll
      for (int hh = 0; hh < 16; ++hh) bf[ni][hh] = bcol[((hh >> 3) << 4) + (hl << 3) + (hh & 7)];
    }
    #pragma unroll
    for (int mi = 0; mi < 2; ++mi)
      #pragma unroll
      for (int ni = 0; ni < 2; ++ni)
        acc[mi][ni] = wmma_f16(af[mi], bf[ni], acc[mi][ni]);

    wait_async_all();
    __syncthreads();
  }

  // ---- epilogue
  #pragma unroll
  for (int mi = 0; mi < 2; ++mi)
    #pragma unroll
    for (int ni = 0; ni < 2; ++ni)
      #pragma unroll
      for (int r = 0; r < 8; ++r) {
        int gm = m0 + wm * 32 + mi * 16 + r + (hl << 3);  // C: M = r + 8*half
        int gn = n0 + wn * 32 + ni * 16 + ln;             // C: N = lane&15
        float cv = acc[mi][ni][r];
        if constexpr (MODE == 0) {
          int b = gm >> 11, s = gm & (S_ - 1);
          int hh = gn >> 6, dh = gn & (DH_ - 1);
          out16[(((size_t)(b * H_ + hh) * S_) + s) * DH_ + dh] = (h16)cv;
        } else {
          out32[(size_t)gm * D_ + gn] = cv + bias[gn];
        }
      }
}

// ------------------------------------------------------- flash attention
// Q,K,V: [B*H, S, DH] f16.  mask: [B, S] f32 (broadcast over queries).
// Computes S^T = K*Q^T (keys in VGPRs, queries in lanes) so online-softmax
// stats are per-lane scalars; ctx accumulated transposed (ctx^T = V^T * P^T).
__global__ __launch_bounds__(256)
void k_attn(const h16* __restrict__ Q, const h16* __restrict__ K,
            const h16* __restrict__ V, const float* __restrict__ mask,
            h16* __restrict__ ctx) {
  constexpr int LK = 72;   // sK row stride (halfs): 144B, 16B aligned
  constexpr int LV = 40;   // sVt row stride (halfs): b16 stores only
  __shared__ h16   sK[2][32 * LK];    // [key][dh]
  __shared__ h16   sVt[2][64 * LV];   // [dh][key]   (transposed V tile)
  __shared__ float smask[2][32];

  const int tid  = threadIdx.x;
  const int lane = tid & 31, w = tid >> 5;
  const int hl   = lane >> 4, ln = lane & 15;

  const int qt = blockIdx.x & 15;        // 16 query tiles of 128
  const int bh = blockIdx.x >> 4;        // 0..31
  const int b  = bh >> 4;
  const int hh = bh & 15;

  const size_t base = (size_t)bh * S_ * DH_;
  const int qrow = qt * 128 + w * 16;    // this wave's 16 queries

  // Q^T B-fragments for dh slices [0,32) and [32,64), loaded once from global
  v16h qtf[2];
  #pragma unroll
  for (int f = 0; f < 2; ++f) {
    const h16* p = Q + base + (size_t)(qrow + ln) * DH_ + f * 32;
    #pragma unroll
    for (int t = 0; t < 16; ++t) qtf[f][t] = p[((t >> 3) << 4) + (hl << 3) + (t & 7)];
  }

  v8f acc[4];                               // ctx^T: dh slices of 16 x queries
  #pragma unroll
  for (int i = 0; i < 4; ++i)
    #pragma unroll
    for (int r = 0; r < 8; ++r) acc[i][r] = 0.f;
  float mrun = -1e30f, lrun = 0.f;
  const float scale = 0.03125f;             // 1/sqrt(1024)

  const int vr = tid >> 3, vcg = (tid & 7) << 3;   // staging coords

  auto stage = [&](int buf, int kt) {
    // K tile 32x64, natural layout: async b128 per thread
    cp16_async(&sK[buf][vr * LK + vcg],
               K + base + (size_t)(kt * 32 + vr) * DH_ + vcg);
    // V tile transposed into [dh][key] (sync)
    const h16* vsrc = V + base + (size_t)(kt * 32 + vr) * DH_ + vcg;
    #pragma unroll
    for (int j = 0; j < 8; ++j) sVt[buf][(vcg + j) * LV + vr] = vsrc[j];
    if (tid < 32) smask[buf][tid] = mask[b * S_ + kt * 32 + tid];
  };

  stage(0, 0);
  wait_async_all();
  __syncthreads();

  constexpr int KT = S_ / 32;
  for (int kt = 0; kt < KT; ++kt) {
    const int buf = kt & 1;
    if (kt + 1 < KT) stage(buf ^ 1, kt + 1);

    // ---- S^T fragments for key groups g*16 .. g*16+15
    v8f pfr[2];
    #pragma unroll
    for (int g = 0; g < 2; ++g) {
      v8f sc;
      #pragma unroll
      for (int r = 0; r < 8; ++r) sc[r] = 0.f;
      #pragma unroll
      for (int f = 0; f < 2; ++f) {
        v16h ka;
        const h16* krow = &sK[buf][(g * 16 + ln) * LK + f * 32];
        #pragma unroll
        for (int v = 0; v < 8; ++v) {
          int kk = ((v >> 2) << 4) + (hl << 3) + ((v & 3) << 1);
          ka[2 * v]     = krow[kk];
          ka[2 * v + 1] = krow[kk + 1];
        }
        sc = wmma_f16(ka, qtf[f], sc);
      }
      #pragma unroll
      for (int r = 0; r < 8; ++r)
        sc[r] = sc[r] * scale + smask[buf][g * 16 + r + (hl << 3)];
      pfr[g] = sc;
    }

    // ---- online softmax: per-lane query stats (combine lane halves)
    float mt = -1e30f;
    #pragma unroll
    for (int r = 0; r < 8; ++r) {
      mt = fmaxf(mt, pfr[0][r]);
      mt = fmaxf(mt, pfr[1][r]);
    }
    mt = fmaxf(mt, __shfl_xor(mt, 16, 32));
    float mnew  = fmaxf(mrun, mt);
    float alpha = __expf(mrun - mnew);
    float lsum = 0.f;
    #pragma unroll
    for (int g = 0; g < 2; ++g)
      #pragma unroll
      for (int r = 0; r < 8; ++r) {
        float e = __expf(pfr[g][r] - mnew);
        pfr[g][r] = e;
        lsum += e;
      }
    lsum += __shfl_xor(lsum, 16, 32);
    lrun = lrun * alpha + lsum;
    mrun = mnew;
    #pragma unroll
    for (int i = 0; i < 4; ++i)
      #pragma unroll
      for (int r = 0; r < 8; ++r) acc[i][r] *= alpha;

    // ---- build P^T B-fragment via cross-half lane exchange
    v16h pb;
    #pragma unroll
    for (int r = 0; r < 8; ++r) {
      float p0o = pfr[0][r], p1o = pfr[1][r];
      float p0x = __shfl_xor(p0o, 16, 32);
      float p1x = __shfl_xor(p1o, 16, 32);
      if (hl == 0) { pb[r] = (h16)p0o; pb[r + 8] = (h16)p0x; }
      else         { pb[r] = (h16)p1x; pb[r + 8] = (h16)p1o; }
    }

    // ---- ctx^T += V^T x P^T
    #pragma unroll
    for (int sl = 0; sl < 4; ++sl) {
      v16h va;
      const h16* vrow = &sVt[buf][(sl * 16 + ln) * LV];
      #pragma unroll
      for (int v = 0; v < 8; ++v) {
        int kk = (hl << 4) + (v << 1);
        va[2 * v]     = vrow[kk];
        va[2 * v + 1] = vrow[kk + 1];
      }
      acc[sl] = wmma_f16(va, pb, acc[sl]);
    }

    wait_async_all();
    __syncthreads();
  }

  // ---- finalize and scatter ctx as [B,S,D] f16 (input of output projection)
  float inv = 1.f / lrun;
  #pragma unroll
  for (int sl = 0; sl < 4; ++sl)
    #pragma unroll
    for (int r = 0; r < 8; ++r) {
      int dh = sl * 16 + r + (hl << 3);
      int q  = qrow + ln;
      ctx[(size_t)(b * S_ + q) * D_ + hh * DH_ + dh] = (h16)(acc[sl][r] * inv);
    }
}

// ----------------------------------------------------------------- launcher
extern "C" void kernel_launch(void* const* d_in, const int* in_sizes, int n_in,
                              void* d_out, int out_size, void* d_ws, size_t ws_size,
                              hipStream_t stream) {
  const float* x    = (const float*)d_in[0];
  const float* mask = (const float*)d_in[1];
  const float* Wq   = (const float*)d_in[2];
  const float* Wk   = (const float*)d_in[3];
  const float* Wv   = (const float*)d_in[4];
  const float* Wo   = (const float*)d_in[5];
  const float* bo   = (const float*)d_in[6];
  float* out = (float*)d_out;

  h16* ws = (h16*)d_ws;
  const size_t MD = (size_t)B_ * S_ * D_;   // 4M elems
  const size_t WD = (size_t)D_ * D_;        // 1M elems
  h16* x16  = ws;
  h16* Wq16 = x16  + MD;
  h16* Wk16 = Wq16 + WD;
  h16* Wv16 = Wk16 + WD;
  h16* Wo16 = Wv16 + WD;
  h16* Q16  = Wo16 + WD;
  h16* K16  = Q16  + MD;
  h16* V16  = K16  + MD;
  h16* C16  = V16  + MD;

  k_cvt<<<1024, 256, 0, stream>>>(x,  x16,  (int)MD);
  k_cvt<<<512,  256, 0, stream>>>(Wq, Wq16, (int)WD);
  k_cvt<<<512,  256, 0, stream>>>(Wk, Wk16, (int)WD);
  k_cvt<<<512,  256, 0, stream>>>(Wv, Wv16, (int)WD);
  k_cvt<<<512,  256, 0, stream>>>(Wo, Wo16, (int)WD);

  const int gGemm = (B_ * S_ / 128) * (D_ / 64);   // 512 blocks
  k_gemm<0><<<gGemm, 256, 0, stream>>>(x16, Wq16, Q16, nullptr, nullptr);
  k_gemm<0><<<gGemm, 256, 0, stream>>>(x16, Wk16, K16, nullptr, nullptr);
  k_gemm<0><<<gGemm, 256, 0, stream>>>(x16, Wv16, V16, nullptr, nullptr);

  k_attn<<<B_ * H_ * (S_ / 128), 256, 0, stream>>>(Q16, K16, V16, mask, C16);

  k_gemm<1><<<gGemm, 256, 0, stream>>>(C16, Wo16, nullptr, out, bo);
}
